// CutCrossEntropyLoss_46042049413851
// MI455X (gfx1250) — compile-verified, matrix operand order checked by reading, not run on hardware
//
#include <hip/hip_runtime.h>
#include <hip/hip_bf16.h>
#include <math.h>

typedef __bf16 bf16_t;
typedef bf16_t v16bf __attribute__((ext_vector_type(16)));
typedef float  v8f   __attribute__((ext_vector_type(8)));
typedef float  v4f   __attribute__((ext_vector_type(4)));
typedef float  v2f   __attribute__((ext_vector_type(2)));
typedef unsigned short v4us __attribute__((ext_vector_type(4)));

#define IGNORE_INDEX (-100)

// ---------------- problem / tiling constants ----------------------------------------
constexpr int Dk        = 1024;            // hidden dim
constexpr int KSTEP     = 32;              // bf16 WMMA K
constexpr int NK        = Dk / KSTEP;      // 32 k-steps
constexpr int ROWS      = 32;              // rows per workgroup (2 row-tiles per wave)
constexpr int VB        = 1024;            // vocab columns per workgroup
constexpr int WAVES     = 8;
constexpr int PASS_COLS = WAVES * 16;      // 128 cols per pass
constexpr int NPASS     = VB / PASS_COLS;  // 8

// LDS A layout: [split(hi/lo)][tile(2)][kk(32)] chunks of 1088B;
// lane L's 16 bf16 fragment elems live at  L*32 + (L>>3)*16  (bank-even padding).
constexpr int CHUNK_B   = 1088;
constexpr int A_SPLIT_B = 2 * NK * CHUNK_B;     // 69632 B per hi/lo split
constexpr int A_BYTES   = 2 * A_SPLIT_B;        // 139264 B
constexpr int MS_FLOATS = WAVES * ROWS * 16 * 2;
constexpr int LDS_BYTES = A_BYTES + MS_FLOATS * 4;  // 172032 B

static __device__ __forceinline__ int lane_off(int L) { return L * 32 + (L >> 3) * 16; }

// truncation-based fp32 -> (bf16 hi, bf16 lo) split
static __device__ __forceinline__ void split_trunc(float x, unsigned short& h, unsigned short& l) {
  unsigned u  = __builtin_bit_cast(unsigned, x);
  unsigned hb = u & 0xFFFF0000u;
  float    r  = x - __builtin_bit_cast(float, hb);
  h = (unsigned short)(u >> 16);
  l = (unsigned short)(__builtin_bit_cast(unsigned, r) >> 16);
}

union BPack { unsigned u[8]; v16bf v; };

// ---------------- Kernel 0 (big-ws path): pre-split weight into bf16 hi/lo planes ----
__global__ __launch_bounds__(256)
void cce_weight_split(const float* __restrict__ w,
                      unsigned short* __restrict__ hi,
                      unsigned short* __restrict__ lo,
                      long long nquads)
{
  long long q = (long long)blockIdx.x * 256 + threadIdx.x;
  if (q >= nquads) return;
  v4f x = *(const v4f*)(w + q * 4);
  v4us hv, lv;
#pragma unroll
  for (int i = 0; i < 4; ++i) {
    unsigned short h, l;
    split_trunc(x[i], h, l);
    hv[i] = h; lv[i] = l;
  }
  *(v4us*)(hi + q * 4) = hv;
  *(v4us*)(lo + q * 4) = lv;
}

// ---------------- Kernel 1: partial logsumexp over a (32 rows x 1024 cols) block ----
// PRESPLIT=1: B fragments loaded directly from bf16 hi/lo planes (no conversion VALU)
// PRESPLIT=0: B fragments converted on the fly from fp32 weight
template <int PRESPLIT>
__global__ __launch_bounds__(256)
void cce_lse_partial(const float* __restrict__ hidden,
                     const float* __restrict__ weight,
                     const unsigned short* __restrict__ wHi,
                     const unsigned short* __restrict__ wLo,
                     const float* __restrict__ bias,
                     float* __restrict__ partials,   // (m,s) pairs: [row][vblock]
                     int NVB)
{
  extern __shared__ char smem[];
  char*  sAhi = smem;
  char*  sAlo = smem + A_SPLIT_B;
  float* sMS  = (float*)(smem + A_BYTES);

  const int tid  = threadIdx.x;
  const int row0 = blockIdx.x * ROWS;
  const long long vb0 = (long long)blockIdx.y * VB;

  // ---- stage hidden strip: split fp32 -> bf16 hi/lo, pre-permuted to fragment order
  for (int i = tid; i < ROWS * Dk; i += 256) {
    int r = i >> 10;
    int d = i & (Dk - 1);
    int tile = r >> 4, mrow = r & 15;
    int kk = d >> 5,  c    = d & 31;
    int h = (c >> 3) & 1;
    int e = (c & 7) | ((c >> 4) << 3);
    int ln = h * 16 + mrow;
    int off = (tile * NK + kk) * CHUNK_B + lane_off(ln) + e * 2;

    unsigned short hb, lb;
    split_trunc(hidden[(size_t)(row0 + r) * Dk + d], hb, lb);
    *(unsigned short*)(sAhi + off) = hb;
    *(unsigned short*)(sAlo + off) = lb;
  }
  __syncthreads();

  const int wave = tid >> 5;
  const int lane = tid & 31;
  const int half = lane >> 4;
  const int l16  = lane & 15;
  const int loff = lane_off(lane);

  float m[2][8], s[2][8];
#pragma unroll
  for (int t = 0; t < 2; ++t)
#pragma unroll
    for (int r = 0; r < 8; ++r) { m[t][r] = -__builtin_inff(); s[t][r] = 0.f; }

  for (int pass = 0; pass < NPASS; ++pass) {
    const long long col = vb0 + (long long)pass * PASS_COLS + wave * 16 + l16;
    const float cbias = bias[col];
    const float*          wrow = weight + (size_t)col * Dk;
    const unsigned short* whr  = wHi + (size_t)col * Dk;
    const unsigned short* wlr  = wLo + (size_t)col * Dk;
    if (pass + 1 < NPASS) {
      if (PRESPLIT) __builtin_prefetch(whr + (size_t)PASS_COLS * Dk, 0, 1);
      else          __builtin_prefetch(wrow + (size_t)PASS_COLS * Dk, 0, 1);
    }

    v8f acc0 = {}, acc1 = {};
    for (int kk = 0; kk < NK; ++kk) {
      const int kA = kk * KSTEP + half * 8;

      // ---- B fragment (shared by both row-tiles)
      BPack bh, bl;
      if (PRESPLIT) {
        __builtin_memcpy(&bh,              whr + kA,      16);
        __builtin_memcpy((char*)&bh + 16,  whr + kA + 16, 16);
        __builtin_memcpy(&bl,              wlr + kA,      16);
        __builtin_memcpy((char*)&bl + 16,  wlr + kA + 16, 16);
      } else {
        v4f w0 = *(const v4f*)(wrow + kA);
        v4f w1 = *(const v4f*)(wrow + kA + 4);
        v4f w2 = *(const v4f*)(wrow + kA + 16);
        v4f w3 = *(const v4f*)(wrow + kA + 20);
        float wv[16];
#pragma unroll
        for (int i = 0; i < 4; ++i) { wv[i] = w0[i]; wv[4+i] = w1[i]; wv[8+i] = w2[i]; wv[12+i] = w3[i]; }
#pragma unroll
        for (int i = 0; i < 8; ++i) {
          float x0 = wv[2*i], x1 = wv[2*i+1];
          unsigned u0 = __builtin_bit_cast(unsigned, x0);
          unsigned u1 = __builtin_bit_cast(unsigned, x1);
          unsigned h0 = u0 & 0xFFFF0000u;
          unsigned h1 = u1 & 0xFFFF0000u;
          bh.u[i] = (u0 >> 16) | h1;
          float r0 = x0 - __builtin_bit_cast(float, h0);
          float r1 = x1 - __builtin_bit_cast(float, h1);
          bl.u[i] = (__builtin_bit_cast(unsigned, r0) >> 16) |
                    (__builtin_bit_cast(unsigned, r1) & 0xFFFF0000u);
        }
      }

      // ---- A fragments: contiguous 32B per lane
      v16bf a0h, a1h, a0l, a1l;
      __builtin_memcpy(&a0h, sAhi + (0 * NK + kk) * CHUNK_B + loff, 32);
      __builtin_memcpy(&a1h, sAhi + (1 * NK + kk) * CHUNK_B + loff, 32);
      __builtin_memcpy(&a0l, sAlo + (0 * NK + kk) * CHUNK_B + loff, 32);
      __builtin_memcpy(&a1l, sAlo + (1 * NK + kk) * CHUNK_B + loff, 32);

      // bf16x3 split-precision: hi*hi + hi*lo + lo*hi
      acc0 = __builtin_amdgcn_wmma_f32_16x16x32_bf16(false, a0h, false, bh.v, (short)0, acc0, false, false);
      acc1 = __builtin_amdgcn_wmma_f32_16x16x32_bf16(false, a1h, false, bh.v, (short)0, acc1, false, false);
      acc0 = __builtin_amdgcn_wmma_f32_16x16x32_bf16(false, a0h, false, bl.v, (short)0, acc0, false, false);
      acc1 = __builtin_amdgcn_wmma_f32_16x16x32_bf16(false, a1h, false, bl.v, (short)0, acc1, false, false);
      acc0 = __builtin_amdgcn_wmma_f32_16x16x32_bf16(false, a0l, false, bh.v, (short)0, acc0, false, false);
      acc1 = __builtin_amdgcn_wmma_f32_16x16x32_bf16(false, a1l, false, bh.v, (short)0, acc1, false, false);
    }

    // online logsumexp: this lane owns one column; rows = tile*16 + half*8 + r
#pragma unroll
    for (int r = 0; r < 8; ++r) {
      float v0  = acc0[r] + cbias;
      float nm0 = fmaxf(m[0][r], v0);
      s[0][r] = s[0][r] * __expf(m[0][r] - nm0) + __expf(v0 - nm0);
      m[0][r] = nm0;
      float v1  = acc1[r] + cbias;
      float nm1 = fmaxf(m[1][r], v1);
      s[1][r] = s[1][r] * __expf(m[1][r] - nm1) + __expf(v1 - nm1);
      m[1][r] = nm1;
    }
  }

  // stash per-lane (m,s); merge across 16 column-slots and 8 waves
#pragma unroll
  for (int t = 0; t < 2; ++t)
#pragma unroll
    for (int r = 0; r < 8; ++r) {
      int row = t * 16 + half * 8 + r;
      int idx = ((wave * ROWS + row) * 16 + l16) * 2;
      sMS[idx]     = m[t][r];
      sMS[idx + 1] = s[t][r];
    }
  __syncthreads();

  if (tid < ROWS) {
    int row = tid;
    float M = -__builtin_inff(), S = 0.f;
    for (int w = 0; w < WAVES; ++w) {
      for (int c = 0; c < 16; ++c) {
        int idx = ((w * ROWS + row) * 16 + c) * 2;
        float mm = sMS[idx], ss = sMS[idx + 1];
        float nm = fmaxf(M, mm);
        S = S * __expf(M - nm) + ss * __expf(mm - nm);
        M = nm;
      }
    }
    v2f out; out.x = M; out.y = S;
    ((v2f*)partials)[(size_t)(row0 + row) * NVB + blockIdx.y] = out;
  }
}

// ---------------- Kernel 2: per-row lse merge + target logit -------------------------
__global__ __launch_bounds__(128)
void cce_row_finalize(const float* __restrict__ hidden,
                      const float* __restrict__ weight,
                      const float* __restrict__ bias,
                      const int* __restrict__ labels,
                      const float* __restrict__ partials,
                      float* __restrict__ ce,
                      float* __restrict__ vld,
                      int NVB, int Dd)
{
  __shared__ float red[128];
  const int row = blockIdx.x;
  const int t   = threadIdx.x;

  v2f p;
  if (t < NVB) p = ((const v2f*)partials)[(size_t)row * NVB + t];
  else { p.x = -__builtin_inff(); p.y = 0.f; }

  red[t] = p.x;
  __syncthreads();
  for (int off = 64; off > 0; off >>= 1) {
    if (t < off) red[t] = fmaxf(red[t], red[t + off]);
    __syncthreads();
  }
  float M = red[0];
  __syncthreads();

  red[t] = p.y * __expf(p.x - M);
  __syncthreads();
  for (int off = 64; off > 0; off >>= 1) {
    if (t < off) red[t] += red[t + off];
    __syncthreads();
  }
  float lse = M + __logf(red[0] + 1e-10f);
  __syncthreads();

  int lab   = labels[row];
  int valid = (lab != IGNORE_INDEX) ? 1 : 0;
  int sl    = valid ? lab : 0;
  const float* hr = hidden + (size_t)row * Dd;
  const float* wr = weight + (size_t)sl  * Dd;
  float d = 0.f;
  for (int i = t; i < Dd; i += 128) d = fmaf(hr[i], wr[i], d);
  red[t] = d;
  __syncthreads();
  for (int off = 64; off > 0; off >>= 1) {
    if (t < off) red[t] += red[t + off];
    __syncthreads();
  }
  if (t == 0) {
    float tgt = red[0] + bias[sl];
    ce[row]  = valid ? (lse - tgt) : 0.f;
    vld[row] = (float)valid;
  }
}

// ---------------- Kernel 3: scalar reduction ----------------------------------------
__global__ __launch_bounds__(1024)
void cce_reduce(const float* __restrict__ ce, const float* __restrict__ vld,
                float* __restrict__ out, int N)
{
  __shared__ float sc[1024];
  __shared__ float sv[1024];
  int t = threadIdx.x;
  float c = 0.f, v = 0.f;
  for (int i = t; i < N; i += 1024) { c += ce[i]; v += vld[i]; }
  sc[t] = c; sv[t] = v;
  __syncthreads();
  for (int off = 512; off > 0; off >>= 1) {
    if (t < off) { sc[t] += sc[t + off]; sv[t] += sv[t + off]; }
    __syncthreads();
  }
  if (t == 0) out[0] = sc[0] / fmaxf(sv[0], 1.0f);
}

extern "C" void kernel_launch(void* const* d_in, const int* in_sizes, int n_in,
                              void* d_out, int out_size, void* d_ws, size_t ws_size,
                              hipStream_t stream) {
  const float* hidden = (const float*)d_in[0];
  const float* weight = (const float*)d_in[1];
  const float* bias   = (const float*)d_in[2];
  const int*   labels = (const int*)d_in[3];

  const int N   = in_sizes[3];          // 2048
  const int V   = in_sizes[2];          // 131072
  const int Dd  = in_sizes[0] / N;      // 1024
  const int NVB = V / VB;               // 128 vocab blocks

  const size_t tailFloats = (size_t)N * NVB * 2 + 2 * (size_t)N;  // partials + ce + vld
  const size_t wElems     = (size_t)V * Dd;
  const size_t planeBytes = wElems * sizeof(unsigned short);      // 256 MB per plane
  const size_t needBig    = 2 * planeBytes + tailFloats * sizeof(float);

  dim3 g1(N / ROWS, NVB);

  if (ws_size >= needBig) {
    // Big-workspace path: pre-split weight once; conversion-free WMMA inner loop.
    unsigned short* wsHi = (unsigned short*)d_ws;
    unsigned short* wsLo = (unsigned short*)((char*)d_ws + planeBytes);
    float* partials = (float*)((char*)d_ws + 2 * planeBytes);
    float* ce       = partials + (size_t)N * NVB * 2;
    float* vld      = ce + N;

    long long nquads = (long long)(wElems / 4);
    int nblk = (int)((nquads + 255) / 256);
    cce_weight_split<<<nblk, 256, 0, stream>>>(weight, wsHi, wsLo, nquads);
    cce_lse_partial<1><<<g1, 256, LDS_BYTES, stream>>>(hidden, weight, wsHi, wsLo, bias, partials, NVB);
    cce_row_finalize<<<N, 128, 0, stream>>>(hidden, weight, bias, labels, partials, ce, vld, NVB, Dd);
    cce_reduce<<<1, 1024, 0, stream>>>(ce, vld, (float*)d_out, N);
  } else {
    // Small-workspace fallback: on-the-fly weight split.
    float* partials = (float*)d_ws;
    float* ce       = partials + (size_t)N * NVB * 2;
    float* vld      = ce + N;

    cce_lse_partial<0><<<g1, 256, LDS_BYTES, stream>>>(hidden, weight, (const unsigned short*)nullptr,
                                                       (const unsigned short*)nullptr, bias, partials, NVB);
    cce_row_finalize<<<N, 128, 0, stream>>>(hidden, weight, bias, labels, partials, ce, vld, NVB, Dd);
    cce_reduce<<<1, 1024, 0, stream>>>(ce, vld, (float*)d_out, N);
  }
}
